// SuperpointTransformerSeg_70420283785929
// MI455X (gfx1250) — matrix operand order repaired and checked by reading, not compiled, not run on hardware
//
#include <hip/hip_runtime.h>
#include <hip/hip_bf16.h>
#include <stdint.h>

typedef _Float16 h16;
typedef __attribute__((ext_vector_type(16))) _Float16 v16h;
typedef __attribute__((ext_vector_type(8)))  float    v8f;
typedef __attribute__((ext_vector_type(4)))  unsigned int u32x4;

union F16Frag { v16h v; u32x4 q[2]; };

#define SEEDS   256
#define HDIM    256
#define NHEADS  8
#define DHEAD   32
#define DFF     1024
#define NCLS    13
#define NLAYERS 2
#define BATCH   16

__device__ __forceinline__ float gelu_exact(float x) {
    return 0.5f * x * (1.0f + erff(x * 0.70710678118654752f));
}

// ---------------------------------------------------------------------------
// Generic WMMA GEMM: out = act( A[M,K] @ W[N,K]^T + bias (+ resid) ), fp16 A/W,
// fp32 accumulate via v_wmma_f32_16x16x32_f16. One wave per 16x16 output tile.
// M, N multiples of 16; K multiple of 32. EXEC uniform per wave.
// ---------------------------------------------------------------------------
__global__ __launch_bounds__(256) void gemm_wmma_kernel(
    const h16* __restrict__ A, const h16* __restrict__ W,
    const float* __restrict__ bias, const float* __restrict__ resid,
    float* __restrict__ out32, h16* __restrict__ out16,
    int M, int N, int K, int act)
{
    int wave   = (blockIdx.x * blockDim.x + threadIdx.x) >> 5;
    int lane   = threadIdx.x & 31;
    int tilesN = N >> 4;
    int total  = (M >> 4) * tilesN;
    if (wave >= total) return;
    int tm = wave / tilesN, tn = wave % tilesN;
    int hs = lane >> 4;        // half-select (0/1)
    int mr = lane & 15;

    const h16* Arow = A + (size_t)(tm * 16 + mr) * K + hs * 8;
    const h16* Brow = W + (size_t)(tn * 16 + mr) * K + hs * 8;

    v8f acc = {};
    for (int k0 = 0; k0 < K; k0 += 32) {
        __builtin_prefetch(Arow + k0 + 64, 0, 3);   // WGP-scope prefetch
        F16Frag au, bu;
        au.q[0] = *(const u32x4*)(Arow + k0);
        au.q[1] = *(const u32x4*)(Arow + k0 + 16);
        bu.q[0] = *(const u32x4*)(Brow + k0);
        bu.q[1] = *(const u32x4*)(Brow + k0 + 16);
        acc = __builtin_amdgcn_wmma_f32_16x16x32_f16(
            false, au.v, false, bu.v, (short)0, acc, false, false);
    }

    int n = tn * 16 + mr;
    float bb = bias ? bias[n] : 0.0f;
    #pragma unroll
    for (int r = 0; r < 8; ++r) {
        int m = tm * 16 + hs * 8 + r;
        float v = acc[r] + bb;
        if (resid) v += resid[(size_t)m * N + n];
        if (act)   v  = gelu_exact(v);
        if (out32) out32[(size_t)m * N + n] = v;
        if (out16) out16[(size_t)m * N + n] = (h16)v;
    }
}

// ---------------------------------------------------------------------------
// Attention scores: per (b,h): scores = Q (256x32) @ K^T / sqrt(32).
// One WMMA (K=32) per 16x16 tile. qkv layout: [B*S, 3H], Q|K|V column blocks.
// ---------------------------------------------------------------------------
__global__ __launch_bounds__(256) void attn_scores_kernel(
    const h16* __restrict__ qkv, h16* __restrict__ attn)
{
    int wave = (blockIdx.x * blockDim.x + threadIdx.x) >> 5;
    int lane = threadIdx.x & 31;
    int bh = wave >> 8;          // 16*16 tiles per (b,h)
    int tt = wave & 255;
    int tm = tt >> 4, tn = tt & 15;
    int b = bh >> 3, hh = bh & 7;
    int hs = lane >> 4, mr = lane & 15;

    const h16* Qb = qkv + (size_t)b * SEEDS * (3 * HDIM) + hh * DHEAD;
    const h16* Kb = Qb + HDIM;

    const h16* Arow = Qb + (size_t)(tm * 16 + mr) * (3 * HDIM) + hs * 8;
    const h16* Brow = Kb + (size_t)(tn * 16 + mr) * (3 * HDIM) + hs * 8;

    F16Frag au, bu;
    au.q[0] = *(const u32x4*)(Arow);
    au.q[1] = *(const u32x4*)(Arow + 16);
    bu.q[0] = *(const u32x4*)(Brow);
    bu.q[1] = *(const u32x4*)(Brow + 16);

    v8f acc = {};
    acc = __builtin_amdgcn_wmma_f32_16x16x32_f16(
        false, au.v, false, bu.v, (short)0, acc, false, false);

    const float scale = 0.17677669529663687f;   // 1/sqrt(32)
    int n = tn * 16 + mr;
    #pragma unroll
    for (int r = 0; r < 8; ++r) {
        int m = tm * 16 + hs * 8 + r;
        attn[((size_t)bh * SEEDS + m) * SEEDS + n] = (h16)(acc[r] * scale);
    }
}

// ---------------------------------------------------------------------------
// Row softmax over attn (in place), rows of length 256. The row is staged into
// LDS with async global->LDS copies (ASYNCcnt path), then reduced.
// ---------------------------------------------------------------------------
__global__ __launch_bounds__(256) void softmax_kernel(h16* __restrict__ attn)
{
    __shared__ h16   rowbuf[256];
    __shared__ float red[256];
    int j = threadIdx.x;
    size_t base = (size_t)blockIdx.x * 256;

    // waves 0..3 (threads 0..127) each async-copy one b32 (two f16) into LDS
    if (j < 128) {
        const h16* gp = attn + base + 2 * j;
        unsigned lds_off = (unsigned)(size_t)(&rowbuf[2 * j]);  // addr[31:0] = LDS offset
        asm volatile("global_load_async_to_lds_b32 %0, %1, off"
                     :: "v"(lds_off), "v"(gp) : "memory");
    }
    asm volatile("s_wait_asynccnt 0x0" ::: "memory");
    __syncthreads();

    float x = (float)rowbuf[j];
    red[j] = x; __syncthreads();
    for (int o = 128; o > 0; o >>= 1) {
        if (j < o) red[j] = fmaxf(red[j], red[j + o]);
        __syncthreads();
    }
    float mx = red[0]; __syncthreads();
    float e = expf(x - mx);
    red[j] = e; __syncthreads();
    for (int o = 128; o > 0; o >>= 1) {
        if (j < o) red[j] += red[j + o];
        __syncthreads();
    }
    attn[base + j] = (h16)(e / red[0]);
}

// ---------------------------------------------------------------------------
// Transpose V out of the packed qkv activations: Vt[(b,h)][n][k] (K-major rows)
// so attn@V can use contiguous b128 B-fragment loads like the weight GEMMs.
// ---------------------------------------------------------------------------
__global__ __launch_bounds__(256) void vtrans_kernel(
    const h16* __restrict__ qkv, h16* __restrict__ vt)
{
    int row = blockIdx.x;            // bh*DHEAD + n, total B*NHEADS*DHEAD
    int k = threadIdx.x;             // 0..SEEDS-1
    int bh = row >> 5, n = row & 31;
    int b = bh >> 3, hh = bh & 7;
    vt[(size_t)row * SEEDS + k] =
        qkv[((size_t)b * SEEDS + k) * (3 * HDIM) + 2 * HDIM + hh * DHEAD + n];
}

// ---------------------------------------------------------------------------
// O = attn (256x256) @ V (256x32) per (b,h), using the K-major Vt buffer.
// ---------------------------------------------------------------------------
__global__ __launch_bounds__(256) void attn_av_kernel(
    const h16* __restrict__ attn, const h16* __restrict__ vt,
    h16* __restrict__ obuf)
{
    int wave = (blockIdx.x * blockDim.x + threadIdx.x) >> 5;
    int lane = threadIdx.x & 31;
    int bh = wave >> 5;          // 16*2 tiles per (b,h)
    int tt = wave & 31;
    int tm = tt >> 1, tn = tt & 1;
    int b = bh >> 3, hh = bh & 7;
    int hs = lane >> 4, mr = lane & 15;

    const h16* Arow = attn + (size_t)bh * SEEDS * SEEDS
                           + (size_t)(tm * 16 + mr) * SEEDS + hs * 8;
    const h16* Brow = vt + (size_t)bh * DHEAD * SEEDS
                         + (size_t)(tn * 16 + mr) * SEEDS + hs * 8;

    v8f acc = {};
    for (int k0 = 0; k0 < SEEDS; k0 += 32) {
        F16Frag au, bu;
        au.q[0] = *(const u32x4*)(Arow + k0);
        au.q[1] = *(const u32x4*)(Arow + k0 + 16);
        bu.q[0] = *(const u32x4*)(Brow + k0);
        bu.q[1] = *(const u32x4*)(Brow + k0 + 16);
        acc = __builtin_amdgcn_wmma_f32_16x16x32_f16(
            false, au.v, false, bu.v, (short)0, acc, false, false);
    }

    int n = hh * DHEAD + tn * 16 + mr;
    #pragma unroll
    for (int r = 0; r < 8; ++r) {
        int m = tm * 16 + hs * 8 + r;
        obuf[((size_t)b * SEEDS + m) * HDIM + n] = (h16)acc[r];
    }
}

// ---------------------------------------------------------------------------
// LayerNorm over rows of 256; writes fp32 + fp16 copies.
// ---------------------------------------------------------------------------
__global__ __launch_bounds__(256) void ln_kernel(
    const float* __restrict__ X, const float* __restrict__ g,
    const float* __restrict__ bt, float* __restrict__ out32,
    h16* __restrict__ out16)
{
    __shared__ float red[256];
    int m = blockIdx.x, j = threadIdx.x;
    float x = X[(size_t)m * 256 + j];
    red[j] = x; __syncthreads();
    for (int o = 128; o > 0; o >>= 1) { if (j < o) red[j] += red[j + o]; __syncthreads(); }
    float mean = red[0] * (1.0f / 256.0f); __syncthreads();
    float d = x - mean;
    red[j] = d * d; __syncthreads();
    for (int o = 128; o > 0; o >>= 1) { if (j < o) red[j] += red[j + o]; __syncthreads(); }
    float var = red[0] * (1.0f / 256.0f);
    float y = d * rsqrtf(var + 1e-5f) * g[j] + bt[j];
    if (out32) out32[(size_t)m * 256 + j] = y;
    if (out16) out16[(size_t)m * 256 + j] = (h16)y;
}

// ---------------------------------------------------------------------------
// Nearest-seed assignment: argmin_s (0.5|s|^2 - x.s) per point.
// ---------------------------------------------------------------------------
__global__ __launch_bounds__(256) void assign_kernel(
    const float* __restrict__ xyz, const int* __restrict__ seed_idx,
    int* __restrict__ assign, int Npts)
{
    __shared__ float sx[256], sy[256], sz[256], sn[256];
    int b = blockIdx.y, t = threadIdx.x;
    int sid = seed_idx[t];
    const float* sp = xyz + ((size_t)b * Npts + sid) * 3;
    float ax = sp[0], ay = sp[1], az = sp[2];
    sx[t] = ax; sy[t] = ay; sz[t] = az;
    sn[t] = 0.5f * (ax * ax + ay * ay + az * az);
    __syncthreads();
    int n = blockIdx.x * 256 + t;
    const float* p = xyz + ((size_t)b * Npts + n) * 3;
    float px = p[0], py = p[1], pz = p[2];
    float best = 3.4e38f; int bi = 0;
    for (int s = 0; s < 256; ++s) {
        float sc = sn[s] - (px * sx[s] + py * sy[s] + pz * sz[s]);
        if (sc < best) { best = sc; bi = s; }
    }
    assign[(size_t)b * Npts + n] = bi;
}

__global__ __launch_bounds__(256) void segsum_kernel(
    const float* __restrict__ xyz, const float* __restrict__ feat,
    const int* __restrict__ assign, float* __restrict__ sum_xyz,
    float* __restrict__ sum_feat, float* __restrict__ cnt, int Npts)
{
    int id = blockIdx.x * 256 + threadIdx.x;
    int b = id / Npts;
    int seg = b * SEEDS + assign[id];
    atomicAdd(&sum_xyz[seg * 3 + 0], xyz[(size_t)id * 3 + 0]);
    atomicAdd(&sum_xyz[seg * 3 + 1], xyz[(size_t)id * 3 + 1]);
    atomicAdd(&sum_xyz[seg * 3 + 2], xyz[(size_t)id * 3 + 2]);
    atomicAdd(&sum_feat[seg * 4 + 0], feat[(size_t)id * 4 + 0]);
    atomicAdd(&sum_feat[seg * 4 + 1], feat[(size_t)id * 4 + 1]);
    atomicAdd(&sum_feat[seg * 4 + 2], feat[(size_t)id * 4 + 2]);
    atomicAdd(&sum_feat[seg * 4 + 3], feat[(size_t)id * 4 + 3]);
    atomicAdd(&cnt[seg], 1.0f);
}

// sp_features (center|feat_mean, 7) -> tokens = spf @ proj_w^T + proj_b
__global__ __launch_bounds__(256) void tokens_proj_kernel(
    const float* __restrict__ xyz, const int* __restrict__ seed_idx,
    const float* __restrict__ sum_xyz, const float* __restrict__ sum_feat,
    const float* __restrict__ cnt, const float* __restrict__ proj_w,
    const float* __restrict__ proj_b, float* __restrict__ tok32,
    h16* __restrict__ tok16, int Npts)
{
    __shared__ float spf[8];
    int m = blockIdx.x;               // 0 .. B*S-1
    int b = m >> 8, s = m & 255;
    if (threadIdx.x == 0) {
        float c = cnt[m];
        if (c == 0.0f) {
            int sid = seed_idx[s];
            const float* p = xyz + ((size_t)b * Npts + sid) * 3;
            spf[0] = p[0]; spf[1] = p[1]; spf[2] = p[2];
            spf[3] = spf[4] = spf[5] = spf[6] = 0.0f;
        } else {
            float inv = 1.0f / c;
            spf[0] = sum_xyz[m * 3 + 0] * inv;
            spf[1] = sum_xyz[m * 3 + 1] * inv;
            spf[2] = sum_xyz[m * 3 + 2] * inv;
            spf[3] = sum_feat[m * 4 + 0] * inv;
            spf[4] = sum_feat[m * 4 + 1] * inv;
            spf[5] = sum_feat[m * 4 + 2] * inv;
            spf[6] = sum_feat[m * 4 + 3] * inv;
        }
    }
    __syncthreads();
    int j = threadIdx.x;
    const float* w = proj_w + j * 7;
    float v = proj_b[j];
    #pragma unroll
    for (int c = 0; c < 7; ++c) v += spf[c] * w[c];
    tok32[(size_t)m * 256 + j] = v;
    tok16[(size_t)m * 256 + j] = (h16)v;
}

// head2: sp_logits[m][c] = gelu_hidden[m] . head2_w[c] + head2_b[c]  (NC=13)
__global__ __launch_bounds__(256) void head2_kernel(
    const h16* __restrict__ hh, const float* __restrict__ w2,
    const float* __restrict__ b2, float* __restrict__ sp_logits, int Mrows)
{
    int id = blockIdx.x * 256 + threadIdx.x;
    if (id >= Mrows * NCLS) return;
    int m = id / NCLS, c = id % NCLS;
    const h16* hr = hh + (size_t)m * 256;
    const float* w = w2 + c * 256;
    float v = b2[c];
    for (int k = 0; k < 256; ++k) v += (float)hr[k] * w[k];
    sp_logits[id] = v;
}

__global__ __launch_bounds__(256) void scatter_kernel(
    const float* __restrict__ sp_logits, const int* __restrict__ assign,
    float* __restrict__ out, int Npts)
{
    int id = blockIdx.x * 256 + threadIdx.x;
    int b = id / Npts;
    int a = assign[id];
    const float* src = sp_logits + ((size_t)b * SEEDS + a) * NCLS;
    float* dst = out + (size_t)id * NCLS;
    #pragma unroll
    for (int c = 0; c < NCLS; ++c) dst[c] = src[c];
}

__global__ __launch_bounds__(256) void cvt_f32_f16_kernel(
    const float* __restrict__ src, h16* __restrict__ dst, int n)
{
    int id = blockIdx.x * 256 + threadIdx.x;
    if (id < n) dst[id] = (h16)src[id];
}

// ---------------------------------------------------------------------------
extern "C" void kernel_launch(void* const* d_in, const int* in_sizes, int n_in,
                              void* d_out, int out_size, void* d_ws, size_t ws_size,
                              hipStream_t stream) {
    const float* xyz       = (const float*)d_in[0];
    const float* features  = (const float*)d_in[1];
    const int*   seed_idx  = (const int*)d_in[2];
    const float* proj_w    = (const float*)d_in[3];
    const float* proj_b    = (const float*)d_in[4];
    const float* qkv_w     = (const float*)d_in[5];
    const float* qkv_b     = (const float*)d_in[6];
    const float* out_w     = (const float*)d_in[7];
    const float* out_b     = (const float*)d_in[8];
    const float* ln1_g     = (const float*)d_in[9];
    const float* ln1_b     = (const float*)d_in[10];
    const float* ln2_g     = (const float*)d_in[11];
    const float* ln2_b     = (const float*)d_in[12];
    const float* ff1_w     = (const float*)d_in[13];
    const float* ff1_b     = (const float*)d_in[14];
    const float* ff2_w     = (const float*)d_in[15];
    const float* ff2_b     = (const float*)d_in[16];
    const float* head_ln_g = (const float*)d_in[17];
    const float* head_ln_b = (const float*)d_in[18];
    const float* head1_w   = (const float*)d_in[19];
    const float* head1_b   = (const float*)d_in[20];
    const float* head2_w   = (const float*)d_in[21];
    const float* head2_b   = (const float*)d_in[22];

    const int B    = BATCH;
    const int Npts = (in_sizes[1] / 4) / B;          // features: B*N*4
    const int BN   = B * Npts;
    const int Mtok = B * SEEDS;                      // 4096 tokens

    // ---- workspace bump allocator ----
    char* base = (char*)d_ws;
    size_t off = 0;
    auto alloc = [&](size_t bytes) -> char* {
        char* p = base + off;
        off = (off + bytes + 255) & ~(size_t)255;
        return p;
    };
    int*   ws_assign = (int*)  alloc((size_t)BN * 4);
    float* ws_cnt    = (float*)alloc((size_t)B * SEEDS * 4);
    float* ws_sxyz   = (float*)alloc((size_t)B * SEEDS * 3 * 4);
    float* ws_sfeat  = (float*)alloc((size_t)B * SEEDS * 4 * 4);
    float* tok32     = (float*)alloc((size_t)Mtok * HDIM * 4);
    h16*   tok16     = (h16*)  alloc((size_t)Mtok * HDIM * 2);
    h16*   qkv16     = (h16*)  alloc((size_t)Mtok * 3 * HDIM * 2);
    h16*   attn16    = (h16*)  alloc((size_t)B * NHEADS * SEEDS * SEEDS * 2);
    h16*   vt16      = (h16*)  alloc((size_t)B * NHEADS * DHEAD * SEEDS * 2);
    h16*   obuf16    = (h16*)  alloc((size_t)Mtok * HDIM * 2);
    float* tmp32     = (float*)alloc((size_t)Mtok * HDIM * 4);
    float* lnx32     = (float*)alloc((size_t)Mtok * HDIM * 4);
    h16*   lnx16     = (h16*)  alloc((size_t)Mtok * HDIM * 2);
    h16*   hbuf16    = (h16*)  alloc((size_t)Mtok * DFF * 2);
    float* sp_logits = (float*)alloc((size_t)Mtok * NCLS * 4);
    h16*   qkvw16    = (h16*)  alloc((size_t)NLAYERS * 3 * HDIM * HDIM * 2);
    h16*   outw16    = (h16*)  alloc((size_t)NLAYERS * HDIM * HDIM * 2);
    h16*   ff1w16    = (h16*)  alloc((size_t)NLAYERS * DFF * HDIM * 2);
    h16*   ff2w16    = (h16*)  alloc((size_t)NLAYERS * HDIM * DFF * 2);
    h16*   h1w16     = (h16*)  alloc((size_t)HDIM * HDIM * 2);

    auto cvt = [&](const float* s, h16* d, int n) {
        cvt_f32_f16_kernel<<<(n + 255) / 256, 256, 0, stream>>>(s, d, n);
    };
    cvt(qkv_w, qkvw16, NLAYERS * 3 * HDIM * HDIM);
    cvt(out_w, outw16, NLAYERS * HDIM * HDIM);
    cvt(ff1_w, ff1w16, NLAYERS * DFF * HDIM);
    cvt(ff2_w, ff2w16, NLAYERS * HDIM * DFF);
    cvt(head1_w, h1w16, HDIM * HDIM);

    // ---- superpoint pooling ----
    hipMemsetAsync(ws_cnt,   0, (size_t)B * SEEDS * 4, stream);
    hipMemsetAsync(ws_sxyz,  0, (size_t)B * SEEDS * 3 * 4, stream);
    hipMemsetAsync(ws_sfeat, 0, (size_t)B * SEEDS * 4 * 4, stream);

    assign_kernel<<<dim3(Npts / 256, B), 256, 0, stream>>>(xyz, seed_idx, ws_assign, Npts);
    segsum_kernel<<<BN / 256, 256, 0, stream>>>(xyz, features, ws_assign,
                                                ws_sxyz, ws_sfeat, ws_cnt, Npts);
    tokens_proj_kernel<<<Mtok, 256, 0, stream>>>(xyz, seed_idx, ws_sxyz, ws_sfeat,
                                                 ws_cnt, proj_w, proj_b,
                                                 tok32, tok16, Npts);

    auto gemm = [&](const h16* A, const h16* W, const float* bias, const float* resid,
                    float* o32, h16* o16, int M, int N, int K, int act) {
        int tiles = (M / 16) * (N / 16);
        gemm_wmma_kernel<<<(tiles + 7) / 8, 256, 0, stream>>>(A, W, bias, resid,
                                                              o32, o16, M, N, K, act);
    };

    // ---- transformer encoder layers ----
    for (int l = 0; l < NLAYERS; ++l) {
        const h16* Wqkv = qkvw16 + (size_t)l * 3 * HDIM * HDIM;
        const h16* Wout = outw16 + (size_t)l * HDIM * HDIM;
        const h16* Wf1  = ff1w16 + (size_t)l * DFF * HDIM;
        const h16* Wf2  = ff2w16 + (size_t)l * HDIM * DFF;

        // QKV projection
        gemm(tok16, Wqkv, qkv_b + l * 3 * HDIM, nullptr, nullptr, qkv16,
             Mtok, 3 * HDIM, HDIM, 0);
        // attention
        attn_scores_kernel<<<(B * NHEADS * 256) / 8, 256, 0, stream>>>(qkv16, attn16);
        vtrans_kernel<<<B * NHEADS * DHEAD, 256, 0, stream>>>(qkv16, vt16);
        softmax_kernel<<<B * NHEADS * SEEDS, 256, 0, stream>>>(attn16);
        attn_av_kernel<<<(B * NHEADS * 32) / 8, 256, 0, stream>>>(attn16, vt16, obuf16);
        // output projection + residual
        gemm(obuf16, Wout, out_b + l * HDIM, tok32, tmp32, nullptr,
             Mtok, HDIM, HDIM, 0);
        ln_kernel<<<Mtok, 256, 0, stream>>>(tmp32, ln1_g + l * HDIM, ln1_b + l * HDIM,
                                            lnx32, lnx16);
        // FFN
        gemm(lnx16, Wf1, ff1_b + l * DFF, nullptr, nullptr, hbuf16,
             Mtok, DFF, HDIM, 1);
        gemm(hbuf16, Wf2, ff2_b + l * HDIM, lnx32, tmp32, nullptr,
             Mtok, HDIM, DFF, 0);
        ln_kernel<<<Mtok, 256, 0, stream>>>(tmp32, ln2_g + l * HDIM, ln2_b + l * HDIM,
                                            tok32, tok16);
    }

    // ---- head ----
    ln_kernel<<<Mtok, 256, 0, stream>>>(tok32, head_ln_g, head_ln_b, lnx32, lnx16);
    gemm(lnx16, h1w16, head1_b, nullptr, nullptr, obuf16, Mtok, HDIM, HDIM, 1);
    head2_kernel<<<(Mtok * NCLS + 255) / 256, 256, 0, stream>>>(obuf16, head2_w,
                                                                head2_b, sp_logits, Mtok);
    // ---- per-point scatter of superpoint logits ----
    scatter_kernel<<<BN / 256, 256, 0, stream>>>(sp_logits, ws_assign,
                                                 (float*)d_out, Npts);
}
